// AttnApproximator_85873576116514
// MI455X (gfx1250) — matrix operand order repaired and checked by reading, not compiled, not run on hardware
//
#include <hip/hip_runtime.h>

// ---------------------------------------------------------------------------
// CDNA5 (gfx1250) wave32 bf16-WMMA implementation of the GQA+RoPE+shift block.
// - All GEMM-shaped stages: v_wmma_f32_16x16x32_bf16, fp32 accumulation.
// - A/K tiles staged via TDM (tensor_load_to_lds + s_wait_tensorcnt),
//   double-buffered so DMA overlaps WMMA compute.
// - B/V tiles stored K-pair-packed so fragments load as contiguous b128.
// ---------------------------------------------------------------------------

typedef __bf16 bf16;
typedef __attribute__((ext_vector_type(16))) __bf16 v16bf;
typedef __attribute__((ext_vector_type(8)))  float  v8f;
typedef __attribute__((ext_vector_type(4)))  unsigned int v4u;
typedef __attribute__((ext_vector_type(8)))  int v8i;
typedef __attribute__((ext_vector_type(4)))  int v4i;

union AF { v16bf v; unsigned int u[8]; };

__device__ __forceinline__ v8f wmma_bf16(const AF& a, const AF& b, v8f c) {
  return __builtin_amdgcn_wmma_f32_16x16x32_bf16(
      false, a.v, false, b.v, (short)0, c, false, false);
}

// pack two bf16 (lo from a, hi from b), lane-matched via v_perm_b32
__device__ __forceinline__ unsigned pack_lo(unsigned a, unsigned b) {
  return __builtin_amdgcn_perm(b, a, 0x05040100u);  // {b[1:0], a[1:0]}
}
__device__ __forceinline__ unsigned pack_hi(unsigned a, unsigned b) {
  return __builtin_amdgcn_perm(b, a, 0x07060302u);  // {b[3:2], a[3:2]}
}

// ----------------------- Tensor Data Mover (TDM) ---------------------------
// 2D tile load: tile_d1 rows of tile_d0 8-byte units, global row pitch
// stride_units (8B units), LDS padding appended per row per flags.
__device__ __forceinline__ void tdm_load_2d(unsigned lds_addr, const void* gaddr,
                                            unsigned flags, unsigned tile_d0,
                                            unsigned tile_d1, unsigned tensor_d0,
                                            unsigned stride_units) {
  unsigned long long ga = (unsigned long long)gaddr;
  v4u g0;
  g0.x = 1u;                                   // count=1 valid descriptor
  g0.y = lds_addr;                             // LDS byte address
  g0.z = (unsigned)(ga & 0xFFFFFFFFu);         // global_addr[31:0]
  g0.w = (unsigned)((ga >> 32) & 0x1FFFFFFu) | (2u << 30);  // [56:32] | type=2
  const unsigned TD1 = 0x40000u;               // large tensor_dim1 (no clamp)
  v8i g1;
  g1[0] = (int)flags;                          // wg_mask=0|data_size|pad cfg
  g1[1] = (int)(tensor_d0 << 16);              // abar=0 | tensor_dim0[15:0]
  g1[2] = (int)((tensor_d0 >> 16) | ((TD1 & 0xFFFFu) << 16));
  g1[3] = (int)((TD1 >> 16) | (tile_d0 << 16));
  g1[4] = (int)tile_d1;                        // tile_dim1 | tile_dim2=0
  g1[5] = (int)stride_units;                   // tensor_dim0_stride[31:0]
  g1[6] = 0;
  g1[7] = 0;
  v4i z4 = {0, 0, 0, 0};
  v8i z8 = {0, 0, 0, 0, 0, 0, 0, 0};
  __builtin_amdgcn_tensor_load_to_lds(g0, g1, z4, z4, z8, 0);
}

// data_size=8B; pad_enable; A tile: 16-dword rows + 4-dword pad (80B pitch)
#define TDM_FLAGS_A ((3u << 16) | (1u << 20) | (3u << 22) | (3u << 25))
// K tile: 64-dword rows + 4-dword pad (272B pitch)
#define TDM_FLAGS_K ((3u << 16) | (1u << 20) | (5u << 22) | (3u << 25))

__device__ __forceinline__ unsigned lds_addr_of(const void* p) {
  return (unsigned)(unsigned long long)p;  // generic addr low 32 = LDS offset
}

// -------------------------- fp32 -> bf16 convert ---------------------------
__global__ void cvt_bf16_kernel(const float* __restrict__ src,
                                bf16* __restrict__ dst, long n) {
  long i = (long)blockIdx.x * blockDim.x + threadIdx.x;
  long stride = (long)gridDim.x * blockDim.x;
  for (; i < n; i += stride) dst[i] = (bf16)src[i];
}

// ------------------------------- RoPE --------------------------------------
__global__ void rope_kernel(const float* __restrict__ src,
                            const int* __restrict__ pos,
                            bf16* __restrict__ dst, int S, int nh, long total) {
  long i = (long)blockIdx.x * blockDim.x + threadIdx.x;
  long stride = (long)gridDim.x * blockDim.x;
  for (; i < total; i += stride) {
    int  j = (int)(i & 63);
    long r = i >> 6;
    int  s = (int)((r / nh) % S);
    int  b = (int)(r / ((long)nh * S));
    float p = (float)pos[(long)b * S + s];
    float inv = __expf(-((float)(2 * j) * (1.0f / 128.0f)) * 9.210340371976184f);
    float ang = p * inv;
    float cs = __cosf(ang), sn = __sinf(ang);
    long base = r * 128;
    float x1 = src[base + j];
    float x2 = src[base + 64 + j];
    dst[base + j]      = (bf16)(x1 * cs - x2 * sn);
    dst[base + 64 + j] = (bf16)(x2 * cs + x1 * sn);
  }
}

// ------------------- bf16 GEMM (templated, double-buffered) ----------------
// C[M,N] = A[M,K] * W[K,N]; tile 128x128, 8 waves (2x4), wave tile 64x32.
template <int K, int LDA, int LDW, int LDC>
__global__ __launch_bounds__(256) void gemm128_kernel(
    const bf16* __restrict__ A, const bf16* __restrict__ W,
    float* __restrict__ C) {
  __shared__ alignas(16) bf16 As[2][128][40];       // TDM fills (32 + 8 pad)
  __shared__ alignas(16) unsigned Bf[2][128][20];   // packed K-pairs (16+pad)
  const int t = threadIdx.x;
  const int lane = t & 31, wave = t >> 5;
  const int lr = lane & 15, half = lane >> 4;
  const long m0 = (long)blockIdx.x * 128;
  const long n0 = (long)blockIdx.y * 128;
  const int wm = (wave >> 2) * 64;
  const int wn = (wave & 3) * 32;
  const v8f vz = {0.f, 0.f, 0.f, 0.f, 0.f, 0.f, 0.f, 0.f};
  v8f acc[4][2];
#pragma unroll
  for (int mt = 0; mt < 4; ++mt)
#pragma unroll
    for (int nt = 0; nt < 2; ++nt) acc[mt][nt] = vz;

  const int kp = t >> 4;        // 0..15 : packed k-pair index
  const int np = (t & 15) * 8;  // 8 n-columns per thread
  const int NT = K / 32;

  // ---- prologue: stage tile 0 into buffer 0 ----
  {
    const uint4 wa = *reinterpret_cast<const uint4*>(
        W + (long)(2 * kp) * LDW + n0 + np);
    const uint4 wb = *reinterpret_cast<const uint4*>(
        W + (long)(2 * kp + 1) * LDW + n0 + np);
    Bf[0][np + 0][kp] = pack_lo(wa.x, wb.x);
    Bf[0][np + 1][kp] = pack_hi(wa.x, wb.x);
    Bf[0][np + 2][kp] = pack_lo(wa.y, wb.y);
    Bf[0][np + 3][kp] = pack_hi(wa.y, wb.y);
    Bf[0][np + 4][kp] = pack_lo(wa.z, wb.z);
    Bf[0][np + 5][kp] = pack_hi(wa.z, wb.z);
    Bf[0][np + 6][kp] = pack_lo(wa.w, wb.w);
    Bf[0][np + 7][kp] = pack_hi(wa.w, wb.w);
  }
  if (wave == 0)
    tdm_load_2d(lds_addr_of(&As[0][0][0]), A + m0 * LDA, TDM_FLAGS_A,
                8, 128, 8, LDA / 4);

  for (int i = 0; i < NT; ++i) {
    const int buf = i & 1;
    const bool more = (i + 1) < NT;
    const int kn = (i + 1) * 32;
    if (wave == 0) __builtin_amdgcn_s_wait_tensorcnt(0);  // tile i DMA done
    __syncthreads();  // staging of tile i visible; prev compute done

    // issue next tile's DMA + global loads (overlap with compute below)
    if (wave == 0 && more)
      tdm_load_2d(lds_addr_of(&As[buf ^ 1][0][0]), A + m0 * LDA + kn,
                  TDM_FLAGS_A, 8, 128, 8, LDA / 4);
    uint4 wa = {}, wb = {};
    if (more) {
      wa = *reinterpret_cast<const uint4*>(
          W + (long)(kn + 2 * kp) * LDW + n0 + np);
      wb = *reinterpret_cast<const uint4*>(
          W + (long)(kn + 2 * kp + 1) * LDW + n0 + np);
    }

    // ---- compute on buffer buf ----
    AF a[4], b[2];
#pragma unroll
    for (int mt = 0; mt < 4; ++mt) {
      const bf16* rp = &As[buf][wm + mt * 16 + lr][0];
#pragma unroll
      for (int vv = 0; vv < 8; ++vv) {
        int kc = half * 16 + 2 * vv;
        a[mt].u[vv] = *reinterpret_cast<const unsigned int*>(rp + kc);
      }
    }
#pragma unroll
    for (int nt = 0; nt < 2; ++nt) {
      const unsigned* bp = &Bf[buf][wn + nt * 16 + lr][half * 8];
#pragma unroll
      for (int vv = 0; vv < 8; ++vv) b[nt].u[vv] = bp[vv];
    }
#pragma unroll
    for (int mt = 0; mt < 4; ++mt)
#pragma unroll
      for (int nt = 0; nt < 2; ++nt)
        acc[mt][nt] = wmma_bf16(a[mt], b[nt], acc[mt][nt]);

    // ---- stage next B tile into the other buffer ----
    if (more) {
      unsigned (*bn)[20] = Bf[buf ^ 1];
      bn[np + 0][kp] = pack_lo(wa.x, wb.x);
      bn[np + 1][kp] = pack_hi(wa.x, wb.x);
      bn[np + 2][kp] = pack_lo(wa.y, wb.y);
      bn[np + 3][kp] = pack_hi(wa.y, wb.y);
      bn[np + 4][kp] = pack_lo(wa.z, wb.z);
      bn[np + 5][kp] = pack_hi(wa.z, wb.z);
      bn[np + 6][kp] = pack_lo(wa.w, wb.w);
      bn[np + 7][kp] = pack_hi(wa.w, wb.w);
    }
  }

  float* cp = C + (m0 + wm + 8 * half) * (long)LDC + n0 + wn + lr;
#pragma unroll
  for (int mt = 0; mt < 4; ++mt)
#pragma unroll
    for (int nt = 0; nt < 2; ++nt)
#pragma unroll
      for (int r = 0; r < 8; ++r)
        cp[(mt * 16 + r) * LDC + nt * 16] = acc[mt][nt][r];
}

// ------------- flash attention (causal GQA, double-buffered) ---------------
__global__ __launch_bounds__(256) void flash_attn_kernel(
    const bf16* __restrict__ qb, const bf16* __restrict__ kbuf,
    const bf16* __restrict__ vbuf, float* __restrict__ attn, int S) {
  const int D = 128, Hh = 32, KVh = 8;
  __shared__ alignas(16) bf16 Ks[2][32][136];      // TDM fills (256B + pad)
  __shared__ alignas(16) unsigned Vf[2][128][20];  // packed V K-pairs
  __shared__ alignas(16) bf16 Ps[8][16][40];
  const int t = threadIdx.x, lane = t & 31, wave = t >> 5;
  const int lr = lane & 15, half = lane >> 4;
  const int q0 = blockIdx.x * 128, h = blockIdx.y, b = blockIdx.z;
  const int kvh = h >> 2;

  // preload Q A-fragments (16 rows per wave, full D=128)
  const int qrow = q0 + wave * 16 + lr;
  const bf16* qp = qb + (((long)b * S + qrow) * Hh + h) * D;
  unsigned int qf[4][8];
#pragma unroll
  for (int c = 0; c < 4; ++c)
#pragma unroll
    for (int vv = 0; vv < 8; ++vv) {
      int kc = c * 32 + half * 16 + 2 * vv;
      qf[c][vv] = *reinterpret_cast<const unsigned int*>(qp + kc);
    }

  const v8f vz = {0.f, 0.f, 0.f, 0.f, 0.f, 0.f, 0.f, 0.f};
  v8f accO[8];
#pragma unroll
  for (int j = 0; j < 8; ++j) accO[j] = vz;
  float rm[8], rs[8];
#pragma unroll
  for (int r = 0; r < 8; ++r) { rm[r] = -3.0e38f; rs[r] = 0.f; }

  const int kp = t >> 4;         // packed key-pair index 0..15
  const int d0 = (t & 15) * 8;   // 8 d-columns per thread
  const int nkb = q0 / 32 + 4;
  const float scale = 0.08838834764831845f;
  const long vstep = (long)KVh * D;

  // ---- prologue: stage key-block 0 into buffer 0 ----
  {
    long base = (((long)b * S + 2 * kp) * KVh + kvh) * D + d0;
    const uint4 va = *reinterpret_cast<const uint4*>(vbuf + base);
    const uint4 vb2 = *reinterpret_cast<const uint4*>(vbuf + base + vstep);
    Vf[0][d0 + 0][kp] = pack_lo(va.x, vb2.x);
    Vf[0][d0 + 1][kp] = pack_hi(va.x, vb2.x);
    Vf[0][d0 + 2][kp] = pack_lo(va.y, vb2.y);
    Vf[0][d0 + 3][kp] = pack_hi(va.y, vb2.y);
    Vf[0][d0 + 4][kp] = pack_lo(va.z, vb2.z);
    Vf[0][d0 + 5][kp] = pack_hi(va.z, vb2.z);
    Vf[0][d0 + 6][kp] = pack_lo(va.w, vb2.w);
    Vf[0][d0 + 7][kp] = pack_hi(va.w, vb2.w);
  }
  if (wave == 0)
    tdm_load_2d(lds_addr_of(&Ks[0][0][0]),
                kbuf + ((long)b * S * KVh + kvh) * D, TDM_FLAGS_K,
                32, 32, 32, (KVh * D * 2) / 8);

  for (int ib = 0; ib < nkb; ++ib) {
    const int buf = ib & 1;
    const bool more = (ib + 1) < nkb;
    const int kb0 = ib * 32;
    if (wave == 0) __builtin_amdgcn_s_wait_tensorcnt(0);
    __syncthreads();

    // issue next block's K DMA + V global loads (overlap with compute)
    if (wave == 0 && more)
      tdm_load_2d(lds_addr_of(&Ks[buf ^ 1][0][0]),
                  kbuf + (((long)b * S + kb0 + 32) * KVh + kvh) * D,
                  TDM_FLAGS_K, 32, 32, 32, (KVh * D * 2) / 8);
    uint4 va = {}, vb2 = {};
    if (more) {
      long base = (((long)b * S + kb0 + 32 + 2 * kp) * KVh + kvh) * D + d0;
      va = *reinterpret_cast<const uint4*>(vbuf + base);
      vb2 = *reinterpret_cast<const uint4*>(vbuf + base + vstep);
    }

    // ---- scores: 16 q-rows x 32 keys ----
    v8f sc[2] = {vz, vz};
#pragma unroll
    for (int c = 0; c < 4; ++c) {
      AF a;
#pragma unroll
      for (int vv = 0; vv < 8; ++vv) a.u[vv] = qf[c][vv];
#pragma unroll
      for (int nt = 0; nt < 2; ++nt) {
        AF bb;
        const int key = nt * 16 + lr;
#pragma unroll
        for (int vv = 0; vv < 8; ++vv) {
          int dd = c * 32 + half * 16 + 2 * vv;
          bb.u[vv] =
              *reinterpret_cast<const unsigned int*>(&Ks[buf][key][dd]);
        }
        sc[nt] = wmma_bf16(a, bb, sc[nt]);
      }
    }

    // ---- online softmax (per-row stats replicated in 16-lane groups) ----
#pragma unroll
    for (int r = 0; r < 8; ++r) {
      const int rowg = q0 + wave * 16 + r + 8 * half;
      float s0 = sc[0][r] * scale;
      float s1 = sc[1][r] * scale;
      if (kb0 + lr > rowg)      s0 = -1.0e30f;
      if (kb0 + 16 + lr > rowg) s1 = -1.0e30f;
      float mb = fmaxf(s0, s1);
#pragma unroll
      for (int o = 1; o < 16; o <<= 1) mb = fmaxf(mb, __shfl_xor(mb, o, 32));
      float mnew = fmaxf(rm[r], mb);
      float alpha = __expf(rm[r] - mnew);
      float p0 = __expf(s0 - mnew);
      float p1 = __expf(s1 - mnew);
      float ps = p0 + p1;
#pragma unroll
      for (int o = 1; o < 16; o <<= 1) ps += __shfl_xor(ps, o, 32);
      rs[r] = rs[r] * alpha + ps;
      rm[r] = mnew;
#pragma unroll
      for (int j = 0; j < 8; ++j) accO[j][r] = accO[j][r] * alpha;
      Ps[wave][r + 8 * half][lr]      = (bf16)p0;
      Ps[wave][r + 8 * half][16 + lr] = (bf16)p1;
    }

    // ---- P(16x32) @ V(32x128) ----
    AF pa;
#pragma unroll
    for (int vv = 0; vv < 8; ++vv) {
      int kc = half * 16 + 2 * vv;
      pa.u[vv] = *reinterpret_cast<const unsigned int*>(&Ps[wave][lr][kc]);
    }
#pragma unroll
    for (int nt = 0; nt < 8; ++nt) {
      AF vf;
      const unsigned* vp = &Vf[buf][nt * 16 + lr][half * 8];
#pragma unroll
      for (int vv = 0; vv < 8; ++vv) vf.u[vv] = vp[vv];
      accO[nt] = wmma_bf16(pa, vf, accO[nt]);
    }

    // ---- stage next V tile into the other buffer ----
    if (more) {
      unsigned (*vn)[20] = Vf[buf ^ 1];
      vn[d0 + 0][kp] = pack_lo(va.x, vb2.x);
      vn[d0 + 1][kp] = pack_hi(va.x, vb2.x);
      vn[d0 + 2][kp] = pack_lo(va.y, vb2.y);
      vn[d0 + 3][kp] = pack_hi(va.y, vb2.y);
      vn[d0 + 4][kp] = pack_lo(va.z, vb2.z);
      vn[d0 + 5][kp] = pack_hi(va.z, vb2.z);
      vn[d0 + 6][kp] = pack_lo(va.w, vb2.w);
      vn[d0 + 7][kp] = pack_hi(va.w, vb2.w);
    }
  }

  // normalize + write [B,S,H,D]
  float* op =
      attn + (((long)b * S + q0 + wave * 16 + 8 * half) * 32 + h) * 128 + lr;
#pragma unroll
  for (int nt = 0; nt < 8; ++nt)
#pragma unroll
    for (int r = 0; r < 8; ++r)
      op[r * 32 * 128 + nt * 16] = accO[nt][r] / rs[r];
}

// ---------------- per-head shift correction: attn += q@Wsq + k@Wsk ---------
__global__ __launch_bounds__(256) void delta_add_kernel(
    const bf16* __restrict__ qb, const bf16* __restrict__ kb,
    const bf16* __restrict__ Wsq, const bf16* __restrict__ Wsk,
    float* __restrict__ attn, int S) {
  const int D = 128, Hh = 32, KVh = 8;
  __shared__ alignas(16) bf16 As[128][40];
  __shared__ alignas(16) unsigned Bf[128][20];
  const int t = threadIdx.x, lane = t & 31, wave = t >> 5;
  const int lr = lane & 15, half = lane >> 4;
  const int s0 = blockIdx.x * 128, h = blockIdx.y, b = blockIdx.z;
  const int wm = (wave >> 2) * 64, wn = (wave & 3) * 32;
  const v8f vz = {0.f, 0.f, 0.f, 0.f, 0.f, 0.f, 0.f, 0.f};
  v8f acc[4][2];
#pragma unroll
  for (int mt = 0; mt < 4; ++mt)
#pragma unroll
    for (int nt = 0; nt < 2; ++nt) acc[mt][nt] = vz;

  const int kp = t >> 4, np = (t & 15) * 8;

  for (int pass = 0; pass < 2; ++pass) {
    const bf16* Ab = pass ? (kb + ((long)b * S * KVh + (h >> 2)) * D)
                          : (qb + ((long)b * S * Hh + h) * D);
    const unsigned ldA = pass ? (unsigned)(KVh * D) : (unsigned)(Hh * D);
    const bf16* Wb = (pass ? Wsk : Wsq) + (long)h * D * D;

    for (int kk = 0; kk < 128; kk += 32) {
      __syncthreads();
      {
        const uint4 wa = *reinterpret_cast<const uint4*>(
            Wb + (long)(kk + 2 * kp) * D + np);
        const uint4 wb = *reinterpret_cast<const uint4*>(
            Wb + (long)(kk + 2 * kp + 1) * D + np);
        Bf[np + 0][kp] = pack_lo(wa.x, wb.x);
        Bf[np + 1][kp] = pack_hi(wa.x, wb.x);
        Bf[np + 2][kp] = pack_lo(wa.y, wb.y);
        Bf[np + 3][kp] = pack_hi(wa.y, wb.y);
        Bf[np + 4][kp] = pack_lo(wa.z, wb.z);
        Bf[np + 5][kp] = pack_hi(wa.z, wb.z);
        Bf[np + 6][kp] = pack_lo(wa.w, wb.w);
        Bf[np + 7][kp] = pack_hi(wa.w, wb.w);
      }
      if (wave == 0) {
        tdm_load_2d(lds_addr_of(&As[0][0]), Ab + (long)s0 * ldA + kk,
                    TDM_FLAGS_A, 8, 128, 8, ldA / 4);
        __builtin_amdgcn_s_wait_tensorcnt(0);
      }
      __syncthreads();

      AF a[4], bfr[2];
#pragma unroll
      for (int mt = 0; mt < 4; ++mt) {
        const bf16* rp = &As[wm + mt * 16 + lr][0];
#pragma unroll
        for (int vv = 0; vv < 8; ++vv) {
          int kc = half * 16 + 2 * vv;
          a[mt].u[vv] = *reinterpret_cast<const unsigned int*>(rp + kc);
        }
      }
#pragma unroll
      for (int nt = 0; nt < 2; ++nt) {
        const unsigned* bp = &Bf[wn + nt * 16 + lr][half * 8];
#pragma unroll
        for (int vv = 0; vv < 8; ++vv) bfr[nt].u[vv] = bp[vv];
      }
#pragma unroll
      for (int mt = 0; mt < 4; ++mt)
#pragma unroll
        for (int nt = 0; nt < 2; ++nt)
          acc[mt][nt] = wmma_bf16(a[mt], bfr[nt], acc[mt][nt]);
    }
  }

  float* cp = attn + (((long)b * S + s0 + wm + 8 * half) * Hh + h) * D + wn + lr;
#pragma unroll
  for (int mt = 0; mt < 4; ++mt)
#pragma unroll
    for (int nt = 0; nt < 2; ++nt)
#pragma unroll
      for (int r = 0; r < 8; ++r)
        cp[(mt * 16 + r) * Hh * D + nt * 16] += acc[mt][nt][r];
}

// ---------------------------------------------------------------------------
extern "C" void kernel_launch(void* const* d_in, const int* in_sizes, int n_in,
                              void* d_out, int out_size, void* d_ws, size_t ws_size,
                              hipStream_t stream) {
  const int B = 2, S = 2048, Dm = 4096, H = 32, KV = 8, D = 128;
  const float* X   = (const float*)d_in[0];
  const int*   pos = (const int*)d_in[1];
  const float* Wq  = (const float*)d_in[2];
  const float* Wk  = (const float*)d_in[3];
  const float* Wv  = (const float*)d_in[4];
  const float* Wo  = (const float*)d_in[5];
  const float* Wsq = (const float*)d_in[6];
  const float* Wsk = (const float*)d_in[7];
  float* out = (float*)d_out;

  char* w = (char*)d_ws;
  auto carve = [&](size_t bytes) -> char* {
    char* p = w;
    w += (bytes + 255) & ~(size_t)255;
    return p;
  };
  bf16* Xb   = (bf16*)carve((size_t)B * S * Dm * sizeof(bf16));
  bf16* Wqb  = (bf16*)carve((size_t)Dm * H * D * sizeof(bf16));
  bf16* Wkb  = (bf16*)carve((size_t)Dm * KV * D * sizeof(bf16));
  bf16* Wvb  = (bf16*)carve((size_t)Dm * KV * D * sizeof(bf16));
  bf16* Wob  = (bf16*)carve((size_t)H * D * Dm * sizeof(bf16));
  bf16* Wsqb = (bf16*)carve((size_t)H * D * D * sizeof(bf16));
  bf16* Wskb = (bf16*)carve((size_t)H * D * D * sizeof(bf16));
  float* Qraw = (float*)carve((size_t)B * S * H * D * sizeof(float));
  float* Kraw = (float*)carve((size_t)B * S * KV * D * sizeof(float));
  float* Vraw = (float*)carve((size_t)B * S * KV * D * sizeof(float));
  bf16* qbf = (bf16*)carve((size_t)B * S * H * D * sizeof(bf16));
  bf16* kbf = (bf16*)carve((size_t)B * S * KV * D * sizeof(bf16));
  bf16* vbf = (bf16*)carve((size_t)B * S * KV * D * sizeof(bf16));
  float* attn  = Qraw;          // reuse: Qraw consumed by RoPE first
  bf16*  attnb = (bf16*)Kraw;   // reuse: Kraw+Vraw consumed before

  cvt_bf16_kernel<<<2048, 256, 0, stream>>>(X,   Xb,   (long)B * S * Dm);
  cvt_bf16_kernel<<<2048, 256, 0, stream>>>(Wq,  Wqb,  (long)Dm * H * D);
  cvt_bf16_kernel<<<1024, 256, 0, stream>>>(Wk,  Wkb,  (long)Dm * KV * D);
  cvt_bf16_kernel<<<1024, 256, 0, stream>>>(Wv,  Wvb,  (long)Dm * KV * D);
  cvt_bf16_kernel<<<2048, 256, 0, stream>>>(Wo,  Wob,  (long)H * D * Dm);
  cvt_bf16_kernel<<<256,  256, 0, stream>>>(Wsq, Wsqb, (long)H * D * D);
  cvt_bf16_kernel<<<256,  256, 0, stream>>>(Wsk, Wskb, (long)H * D * D);

  gemm128_kernel<4096, 4096, 4096, 4096>
      <<<dim3(B * S / 128, 32), 256, 0, stream>>>(Xb, Wqb, Qraw);
  gemm128_kernel<4096, 4096, 1024, 1024>
      <<<dim3(B * S / 128, 8), 256, 0, stream>>>(Xb, Wkb, Kraw);
  gemm128_kernel<4096, 4096, 1024, 1024>
      <<<dim3(B * S / 128, 8), 256, 0, stream>>>(Xb, Wvb, Vraw);

  rope_kernel<<<2048, 256, 0, stream>>>(Qraw, pos, qbf, S, H,
                                        (long)B * S * H * (D / 2));
  rope_kernel<<<1024, 256, 0, stream>>>(Kraw, pos, kbf, S, KV,
                                        (long)B * S * KV * (D / 2));
  cvt_bf16_kernel<<<1024, 256, 0, stream>>>(Vraw, vbf, (long)B * S * KV * D);

  flash_attn_kernel<<<dim3(S / 128, H, B), 256, 0, stream>>>(qbf, kbf, vbf,
                                                             attn, S);
  delta_add_kernel<<<dim3(S / 128, H, B), 256, 0, stream>>>(qbf, kbf, Wsqb,
                                                            Wskb, attn, S);

  cvt_bf16_kernel<<<2048, 256, 0, stream>>>(attn, attnb, (long)B * S * H * D);
  gemm128_kernel<4096, 4096, 4096, 4096>
      <<<dim3(B * S / 128, 32), 256, 0, stream>>>(attnb, Wob, out);
}
